// MemoryKernelModel_8761733284285
// MI455X (gfx1250) — matrix operand order repaired and checked by reading, not compiled, not run on hardware
//
#include <hip/hip_runtime.h>
#include <math.h>

typedef __attribute__((ext_vector_type(16))) _Float16 v16h;
typedef __attribute__((ext_vector_type(2)))  __fp16   hf2;   // matches cvt_pkrtz return
typedef __attribute__((ext_vector_type(8)))  float    v8f;
typedef __attribute__((ext_vector_type(8)))  unsigned v8u;
typedef __attribute__((ext_vector_type(4)))  float    f4;

#define B_TOT 8192
#define HHIST 512
#define CCH   8
#define SWIN  32
#define HID   64
#define NSTEP 400
#define DTC   0.05f

__device__ __forceinline__ unsigned pk2(float a, float b) {
    hf2 h = __builtin_amdgcn_cvt_pkrtz(a, b);
    return __builtin_bit_cast(unsigned, h);
}
__device__ __forceinline__ v16h as_h(v8u x) { return __builtin_bit_cast(v16h, x); }

__device__ __forceinline__ v8f wmma16(v8u a, v8u b, v8f c) {
    // D = A(16x32 f16) * B(32x16 f16) + C(16x16 f32)
    return __builtin_amdgcn_wmma_f32_16x16x32_f16(
        false, as_h(a), false, as_h(b), (short)0, c, false, false);
}

// Branchless gelu: 0.5*x*(1+erf(x/sqrt2)), erf via A&S 7.1.26 (|err|<1.5e-7).
// No divergent branches: rcp + native exp + bfi(sign).
__device__ __forceinline__ float gelu_fast(float x) {
    float a = fabsf(x) * 0.70710678118654752f;
    float t = __builtin_amdgcn_rcpf(fmaf(0.3275911f, a, 1.0f));
    float p = fmaf(t, 1.061405429f, -1.453152027f);
    p = fmaf(t, p, 1.421413741f);
    p = fmaf(t, p, -0.284496736f);
    p = fmaf(t, p, 0.254829592f);
    p = p * t;
    float e = __expf(-a * a);                 // native v_exp_f32 path, branchless
    float erfa = fmaf(-p, e, 1.0f);           // erf(|x|/sqrt2)
    float s = copysignf(erfa, x);             // v_bfi
    return 0.5f * x * (1.0f + s);
}

// ---------------------------------------------------------------------------
// Prep: fold rfft -> complex mix -> irfft@t=31 into K[o][c*32+t] (8 x 256)
// c2r convention: imag of DC & Nyquist ignored; middle bins doubled.
// ---------------------------------------------------------------------------
__global__ void k_prep(const float* __restrict__ fwr, const float* __restrict__ fwi,
                       float* __restrict__ Kmat) {
    int i = blockIdx.x * blockDim.x + threadIdx.x;
    if (i >= 8 * 256) return;
    int o = i >> 8;
    int rem = i & 255;
    int c = rem >> 5;
    int t = rem & 31;
    int d = 31 - t;                       // phase index (T - t), T = 31
    const float TWO_PI_OVER_S = 6.283185307179586f / 32.0f;
    const float* fr = fwr + (o * 8 + c) * 17;
    const float* fi = fwi + (o * 8 + c) * 17;
    float acc = fr[0];                    // DC: real part only
    for (int f = 1; f < 16; ++f) {
        float ph = TWO_PI_OVER_S * (float)(f * d);
        float sn = sinf(ph), cs = cosf(ph);
        acc += 2.0f * (fr[f] * cs - fi[f] * sn);
    }
    acc += ((d & 1) ? -1.0f : 1.0f) * fr[16];   // Nyquist: real part only
    Kmat[o * 256 + c * 32 + t] = acc * (1.0f / 32.0f);
}

// ---------------------------------------------------------------------------
// Main scan: one wave32 = 16 batch rows; all state in registers.
//   lanes 0-15  : batch n = base+lane   (M-halves 0-7 of C/D, K-halves 0-15 of B)
//   lanes 16-31 : same batches, upper K/M halves
// ---------------------------------------------------------------------------
__global__ void __launch_bounds__(256, 1)
k_scan(const float* __restrict__ hist_raw, const float* __restrict__ Kmat,
       const float* __restrict__ w1, const float* __restrict__ b1,
       const float* __restrict__ w2, const float* __restrict__ b2,
       float* __restrict__ out) {
    const int lane  = threadIdx.x & 31;
    const int wid   = (int)blockIdx.x * (blockDim.x >> 5) + (threadIdx.x >> 5);
    const int bbase = wid * 16;
    const bool lo   = lane < 16;
    const int  mrow = lane & 15;
    const v8u  zu   = {0, 0, 0, 0, 0, 0, 0, 0};

    // ---- K matrix in WMMA A-layout (8 chunks; chunk q == channel q) ----
    v8u Ka[8];
#pragma unroll
    for (int q = 0; q < 8; ++q) {
        Ka[q] = zu;
        if (mrow < 8) {
            const float* r = Kmat + mrow * 256 + q * 32 + (lo ? 0 : 16);
            f4 a0 = *(const f4*)(r + 0),  a1 = *(const f4*)(r + 4);
            f4 a2 = *(const f4*)(r + 8), a3 = *(const f4*)(r + 12);
            Ka[q][0] = pk2(a0.x, a0.y); Ka[q][1] = pk2(a0.z, a0.w);
            Ka[q][2] = pk2(a1.x, a1.y); Ka[q][3] = pk2(a1.z, a1.w);
            Ka[q][4] = pk2(a2.x, a2.y); Ka[q][5] = pk2(a2.z, a2.w);
            Ka[q][6] = pk2(a3.x, a3.y); Ka[q][7] = pk2(a3.z, a3.w);
        }
    }

    // ---- w1^T tiles (hiddenT rows m*16.., K = input channel, K>=8 zero) ----
    v8u W1[4];
#pragma unroll
    for (int m = 0; m < 4; ++m) {
        W1[m] = zu;
        if (lo) {
            int h = m * 16 + mrow;
            W1[m][0] = pk2(w1[0 * 64 + h], w1[1 * 64 + h]);
            W1[m][1] = pk2(w1[2 * 64 + h], w1[3 * 64 + h]);
            W1[m][2] = pk2(w1[4 * 64 + h], w1[5 * 64 + h]);
            W1[m][3] = pk2(w1[6 * 64 + h], w1[7 * 64 + h]);
        }
    }

    // ---- w2^T tiles (M = out channel <8, K = hidden 0..63 in 2 chunks) ----
    v8u W2[2];
#pragma unroll
    for (int k = 0; k < 2; ++k) {
        W2[k] = zu;
        if (mrow < 8) {
#pragma unroll
            for (int j = 0; j < 8; ++j) {
                int s0 = 2 * j, s1 = 2 * j + 1;
                int kl0 = s0 + (lo ? 0 : 16);
                int kl1 = s1 + (lo ? 0 : 16);
                W2[k][j] = pk2(w2[(k * 32 + kl0) * 8 + mrow],
                               w2[(k * 32 + kl1) * 8 + mrow]);
            }
        }
    }

    // ---- b1 as C-operand tiles; b2 per-VGPR ----
    v8f Cb1[4];
#pragma unroll
    for (int m = 0; m < 4; ++m) {
        const float* bp = b1 + m * 16 + (lo ? 0 : 8);
#pragma unroll
        for (int v = 0; v < 8; ++v) Cb1[m][v] = bp[v];
    }
    float b2r[8];
#pragma unroll
    for (int v = 0; v < 8; ++v) b2r[v] = b2[v];

    // ---- init history window (linear-interp sample of 32 points) ----
    const float* hr = hist_raw + (size_t)(bbase + mrow) * (HHIST * CCH);
    v8u HB[8];
#pragma unroll
    for (int q = 0; q < 8; ++q) HB[q] = zu;
#pragma unroll
    for (int tl = 0; tl < 16; ++tl) {
        int t = (lo ? 0 : 16) + tl;
        float idx = (511.0f * (float)t) * (1.0f / 31.0f);
        float fidx = floorf(idx);
        int fi = (int)fidx;
        int ci = fi + 1; if (ci > 511) ci = 511;
        float w = idx - fidx;
        f4 f0 = *(const f4*)(hr + fi * 8);
        f4 f1 = *(const f4*)(hr + fi * 8 + 4);
        f4 c0 = *(const f4*)(hr + ci * 8);
        f4 c1 = *(const f4*)(hr + ci * 8 + 4);
        float vch[8];
        vch[0] = f0.x + w * (c0.x - f0.x); vch[1] = f0.y + w * (c0.y - f0.y);
        vch[2] = f0.z + w * (c0.z - f0.z); vch[3] = f0.w + w * (c0.w - f0.w);
        vch[4] = f1.x + w * (c1.x - f1.x); vch[5] = f1.y + w * (c1.y - f1.y);
        vch[6] = f1.z + w * (c1.z - f1.z); vch[7] = f1.w + w * (c1.w - f1.w);
#pragma unroll
        for (int c = 0; c < 8; ++c) {
            unsigned hv = pk2(vch[c], 0.0f) & 0xffffu;
            int d = tl >> 1;
            if (tl & 1) HB[c][d] = (HB[c][d] & 0x0000ffffu) | (hv << 16);
            else        HB[c][d] = (HB[c][d] & 0xffff0000u) | hv;
        }
    }

    // ---- u0 = hist0[:, 31, :] == hist_raw[:, 511, :] exactly ----
    float u[8];
    {
        f4 u0 = *(const f4*)(hr + 511 * 8);
        f4 u1 = *(const f4*)(hr + 511 * 8 + 4);
        u[0] = u0.x; u[1] = u0.y; u[2] = u0.z; u[3] = u0.w;
        u[4] = u1.x; u[5] = u1.y; u[6] = u1.z; u[7] = u1.w;
    }

    // packed f16 view of u (lane-local) + cross-half copy; refreshed each step
    unsigned pu[4], spu[4];
#pragma unroll
    for (int j = 0; j < 4; ++j) {
        pu[j]  = pk2(u[2 * j], u[2 * j + 1]);
        spu[j] = (unsigned)__shfl_xor((int)pu[j], 16, 32);
    }

    float* outp = out + (size_t)(bbase + lane) * NSTEP * CCH;

    for (int s = 0; s < NSTEP; ++s) {
        // ---- kterm: two independent 4-deep WMMA chains ----
        v8f accA = {0, 0, 0, 0, 0, 0, 0, 0};
        v8f accB = {0, 0, 0, 0, 0, 0, 0, 0};
#pragma unroll
        for (int q = 0; q < 4; ++q) accA = wmma16(Ka[q], HB[q], accA);
#pragma unroll
        for (int q = 4; q < 8; ++q) accB = wmma16(Ka[q], HB[q], accB);

        // ---- MLP input: u^T as B operand (K = channel, rest zero) ----
        v8u Bu = zu;
        Bu[0] = lo ? pu[0] : 0u;
        Bu[1] = lo ? pu[1] : 0u;
        Bu[2] = lo ? pu[2] : 0u;
        Bu[3] = lo ? pu[3] : 0u;

        // ---- hidden (two halves of 32 channels), gelu, second GEMM ----
#pragma unroll
        for (int half = 0; half < 2; ++half) {
            v8f h0 = wmma16(W1[half * 2 + 0], Bu, Cb1[half * 2 + 0]);
            v8f h1 = wmma16(W1[half * 2 + 1], Bu, Cb1[half * 2 + 1]);
            unsigned p0[4], p1[4];
#pragma unroll
            for (int j = 0; j < 4; ++j) {
                p0[j] = pk2(gelu_fast(h0[2 * j]), gelu_fast(h0[2 * j + 1]));
                p1[j] = pk2(gelu_fast(h1[2 * j]), gelu_fast(h1[2 * j + 1]));
            }
            v8u BB;
#pragma unroll
            for (int j = 0; j < 4; ++j) {
                unsigned sp0 = (unsigned)__shfl_xor((int)p0[j], 16, 32);
                unsigned sp1 = (unsigned)__shfl_xor((int)p1[j], 16, 32);
                BB[j]     = lo ? p0[j] : sp1;
                BB[4 + j] = lo ? sp0   : p1[j];
            }
            if (half == 0) accA = wmma16(W2[0], BB, accA);   // chain into kterm A
            else           accB = wmma16(W2[1], BB, accB);   // chain into kterm B
        }

        // ---- Euler update (exact f32) ----
#pragma unroll
        for (int v = 0; v < 8; ++v)
            u[v] = u[v] + DTC * ((accA[v] + accB[v]) + b2r[v]);

        // ---- refresh packed u + cross-half copy (reused next iter) ----
#pragma unroll
        for (int j = 0; j < 4; ++j) {
            pu[j]  = pk2(u[2 * j], u[2 * j + 1]);
            spu[j] = (unsigned)__shfl_xor((int)pu[j], 16, 32);
        }

        // ---- emit trajectory sample (lanes 0-15 only) ----
        if (lo) {
            f4 s0 = {u[0], u[1], u[2], u[3]};
            f4 s1 = {u[4], u[5], u[6], u[7]};
            *(f4*)(outp + s * 8)     = s0;
            *(f4*)(outp + s * 8 + 4) = s1;
        }

        // ---- shift window one step; inject new u at t=31 ----
#pragma unroll
        for (int q = 0; q < 8; ++q) {
            v8u hq = HB[q];
            // lanes<16 boundary: old t=16 lives in partner lane's dword 0
            unsigned a = (unsigned)__shfl_xor((int)hq[0], 16, 32);
            // lanes>=16 boundary: inject f16(u[q]) of owning lane
            unsigned hv = (q & 1) ? (spu[q >> 1] >> 16) : (spu[q >> 1] & 0xffffu);
            unsigned x = lo ? a : hv;
            v8u nq;
#pragma unroll
            for (int j = 0; j < 7; ++j) nq[j] = (hq[j] >> 16) | (hq[j + 1] << 16);
            nq[7] = (hq[7] >> 16) | (x << 16);
            HB[q] = nq;
        }
    }
}

extern "C" void kernel_launch(void* const* d_in, const int* in_sizes, int n_in,
                              void* d_out, int out_size, void* d_ws, size_t ws_size,
                              hipStream_t stream) {
    (void)in_sizes; (void)n_in; (void)out_size; (void)ws_size;
    const float* hist_raw = (const float*)d_in[0];
    const float* fwr      = (const float*)d_in[1];
    const float* fwi      = (const float*)d_in[2];
    const float* w1       = (const float*)d_in[3];
    const float* b1       = (const float*)d_in[4];
    const float* w2       = (const float*)d_in[5];
    const float* b2       = (const float*)d_in[6];
    float* Kmat = (float*)d_ws;   // 8*256 floats = 8 KB

    k_prep<<<8, 256, 0, stream>>>(fwr, fwi, Kmat);
    // 8192 batches / 16 per wave = 512 waves; 8 waves per 256-thread block
    k_scan<<<64, 256, 0, stream>>>(hist_raw, Kmat, w1, b1, w2, b2, (float*)d_out);
}
